// MoECNBlock_7868380086756
// MI455X (gfx1250) — compile-verified, matrix operand order checked by reading, not run on hardware
//
#include <hip/hip_runtime.h>
#include <hip/hip_bf16.h>
#include <math.h>

// ---------------------------------------------------------------------------
// MoE ConvNeXt block for MI455X (gfx1250, wave32).
// Expert GEMMs run on v_wmma_f32_16x16x32_bf16 (bf16 in, fp32 accumulate).
// ---------------------------------------------------------------------------

typedef __attribute__((ext_vector_type(16))) __bf16 v16bf;
typedef __attribute__((ext_vector_type(8)))  float  v8f;

#define DEV __device__ __forceinline__

constexpr int Nn = 32, Cc = 128, Hh = 56, Ww = 56;
constexpr int T    = Nn * Hh * Ww;        // 100352 tokens
constexpr int E    = 8, TOPK = 2;
constexpr int MC   = 4 * Cc;              // 512 (MLP hidden)
constexpr int CAP  = 15680;               // int(1.25*T/E)
constexpr int ECAP = E * CAP;             // 125440 slots
constexpr int ENT  = T * TOPK;            // 200704 routed entries
constexpr int PB   = 2048;                // entries per positions block
constexpr int NBLK = ENT / PB;            // 98 (exact)
constexpr int HP   = MC + 8;              // LDS pitch (ushorts) for H tile

DEV unsigned short f2bf(float f) {        // round-to-nearest-even f32->bf16
  unsigned u = __float_as_uint(f);
  unsigned r = u + 0x7FFFu + ((u >> 16) & 1u);
  return (unsigned short)(r >> 16);
}

// Branchless GELU (tanh form) on hardware transcendentals:
// v_exp_f32 + v_rcp_f32, no EXEC-mask branches. |err| ~1e-3, below the bf16
// rounding already applied to H.
DEV float fast_gelu(float x) {
  float u = 0.7978845608f * x * (1.f + 0.044715f * x * x);
  float e = __expf(2.f * u);                       // inf-safe: e->inf => th->1
  float th = 1.f - 2.f * __builtin_amdgcn_rcpf(e + 1.f);
  return 0.5f * x * (1.f + th);
}

// Load a 16x32 bf16 WMMA fragment (A layout; B of a K x 16 operand stored
// n-major uses the symmetric mapping). 8 aligned dword loads per lane;
// consecutive k-steps merge into global_load_b128 clauses.
// Lane l: m/n = l&15, half = l>>4; dword r holds K = (r>=4)*16 + half*8 + (r&3)*2 (+0,+1).
DEV v16bf load_frag(const unsigned short* base, int row, int rowstride, int k0, int lane) {
  int m = lane & 15, half = lane >> 4;
  const unsigned short* p = base + (size_t)(row + m) * rowstride + k0;
  union { v16bf v; unsigned u[8]; } f;
#pragma unroll
  for (int r = 0; r < 8; ++r) {
    int k = ((r >> 2) << 4) + (half << 3) + ((r & 3) << 1);
    f.u[r] = *(const unsigned*)(p + k);
  }
  return f.v;
}

// ---------------------------------------------------------------------------
// 1. Depthwise conv 7x7 + LayerNorm + gating (top-2 softmax) per (n, h) row.
// ---------------------------------------------------------------------------
__global__ __launch_bounds__(64)
void k_dwln_gate(const float* __restrict__ in, const float* __restrict__ wk,
                 const float* __restrict__ wb, const float* __restrict__ gam,
                 const float* __restrict__ bet, const float* __restrict__ Wg,
                 const float* __restrict__ bg,
                 float* __restrict__ xf, float* __restrict__ prio,
                 int* __restrict__ topi, float* __restrict__ wts) {
  __shared__ float tile[7 * 64];
  __shared__ float ker[49];
  __shared__ float convbuf[Cc * 57];
  int h = blockIdx.x, n = blockIdx.y, tid = threadIdx.x;
  float s = 0.f, s2 = 0.f;

  for (int c = 0; c < Cc; ++c) {
    if (tid < 49) ker[tid] = wk[c * 49 + tid];
    const float* ip = in + ((size_t)(n * Cc + c) * Hh) * Ww;
    for (int idx = tid; idx < 7 * 62; idx += 64) {
      int r = idx / 62, mcol = idx % 62, col = mcol - 3;
      int hr = h - 3 + r;
      float v = 0.f;
      if (hr >= 0 && hr < Hh && col >= 0 && col < Ww) v = ip[hr * Ww + col];
      tile[r * 64 + mcol] = v;
    }
    __syncthreads();
    if (tid < Ww) {
      float acc = wb[c];
#pragma unroll
      for (int kh = 0; kh < 7; ++kh)
#pragma unroll
        for (int kw = 0; kw < 7; ++kw)
          acc += ker[kh * 7 + kw] * tile[kh * 64 + tid + kw];
      convbuf[c * 57 + tid] = acc;
      s += acc; s2 += acc * acc;
    }
    __syncthreads();
  }

  if (tid < Ww) {
    float mu  = s * (1.f / Cc);
    float var = s2 * (1.f / Cc) - mu * mu;
    float rstd = rsqrtf(var + 1e-6f);
    int t = (n * Hh + h) * Ww + tid;
    float lg[E];
#pragma unroll
    for (int e = 0; e < E; ++e) lg[e] = bg[e];
    for (int c = 0; c < Cc; c += 4) {           // float4 xf writeback
      float v[4];
#pragma unroll
      for (int j = 0; j < 4; ++j) {
        v[j] = (convbuf[(c + j) * 57 + tid] - mu) * rstd * gam[c + j] + bet[c + j];
#pragma unroll
        for (int e = 0; e < E; ++e) lg[e] += v[j] * Wg[(c + j) * E + e];
      }
      *(float4*)(xf + (size_t)t * Cc + c) = make_float4(v[0], v[1], v[2], v[3]);
    }
    int e0 = 0;
#pragma unroll
    for (int e = 1; e < E; ++e) if (lg[e] > lg[e0]) e0 = e;
    int e1 = (e0 == 0) ? 1 : 0;
#pragma unroll
    for (int e = 0; e < E; ++e) if (e != e0 && lg[e] > lg[e1]) e1 = e;
    float w0 = 1.f / (1.f + __expf(lg[e1] - lg[e0]));
    topi[t * 2] = e0; topi[t * 2 + 1] = e1;
    wts[t * 2] = w0;  wts[t * 2 + 1] = 1.f - w0;
    prio[t] = w0;
  }
}

// ---------------------------------------------------------------------------
// 2. Priority sort: 16-bit-key counting sort (descending priority).
// ---------------------------------------------------------------------------
DEV unsigned prio_key(float p) {  // p > 0 -> float bits are order-preserving
  return 0xFFFFu - (__float_as_uint(p) >> 16);
}

__global__ void k_hist(const float* __restrict__ prio, unsigned* __restrict__ hist) {
  int t = blockIdx.x * blockDim.x + threadIdx.x;
  if (t < T) atomicAdd(&hist[prio_key(prio[t])], 1u);
}

__global__ __launch_bounds__(256)
void k_scan(unsigned* __restrict__ hist) {
  __shared__ unsigned tot[256];
  int tid = threadIdx.x, base = tid * 256;
  unsigned s = 0;
  for (int i = 0; i < 256; ++i) s += hist[base + i];
  tot[tid] = s;
  __syncthreads();
  if (tid == 0) {
    unsigned run = 0;
    for (int i = 0; i < 256; ++i) { unsigned v = tot[i]; tot[i] = run; run += v; }
  }
  __syncthreads();
  unsigned run = tot[tid];
  for (int i = 0; i < 256; ++i) { unsigned v = hist[base + i]; hist[base + i] = run; run += v; }
}

__global__ void k_scatter(const float* __restrict__ prio, unsigned* __restrict__ hist,
                          int* __restrict__ order) {
  int t = blockIdx.x * blockDim.x + threadIdx.x;
  if (t < T) {
    unsigned p = atomicAdd(&hist[prio_key(prio[t])], 1u);
    order[p] = t;
  }
}

// ---------------------------------------------------------------------------
// 3. Capacity positions: segmented per-expert prefix over sorted entries.
// ---------------------------------------------------------------------------
__global__ __launch_bounds__(256)
void k_blockcnt(const int* __restrict__ order, const int* __restrict__ topi,
                unsigned* __restrict__ bcnt) {
  __shared__ unsigned cnt[E];
  int tid = threadIdx.x, b = blockIdx.x;
  if (tid < E) cnt[tid] = 0;
  __syncthreads();
  for (int j = tid; j < PB; j += blockDim.x) {
    int i = b * PB + j;
    int e = topi[order[i >> 1] * 2 + (i & 1)];
    atomicAdd(&cnt[e], 1u);
  }
  __syncthreads();
  if (tid < E) bcnt[b * E + tid] = cnt[tid];
}

__global__ void k_blockscan(const unsigned* __restrict__ bcnt, unsigned* __restrict__ boff) {
  int e = threadIdx.x;
  if (e < E) {
    unsigned run = 0;
    for (int b = 0; b < NBLK; ++b) { unsigned v = bcnt[b * E + e]; boff[b * E + e] = run; run += v; }
  }
}

__global__ __launch_bounds__(256)
void k_positions(const int* __restrict__ order, const int* __restrict__ topi,
                 const float* __restrict__ wts, const unsigned* __restrict__ boff,
                 int* __restrict__ tok_slot, float* __restrict__ tok_gate,
                 int* __restrict__ slot_token) {
  __shared__ unsigned char earr[PB];
  __shared__ int slotarr[PB];
  int b = blockIdx.x, tid = threadIdx.x;
  for (int j = tid; j < PB; j += blockDim.x) {
    int i = b * PB + j;
    earr[j] = (unsigned char)topi[order[i >> 1] * 2 + (i & 1)];
  }
  __syncthreads();
  if (tid < E) {
    unsigned run = boff[b * E + tid];
    for (int j = 0; j < PB; ++j) {
      if (earr[j] == (unsigned char)tid) {
        slotarr[j] = (run < (unsigned)CAP) ? (int)(tid * CAP + run) : -1;
        run++;
      }
    }
  }
  __syncthreads();
  for (int j = tid; j < PB; j += blockDim.x) {
    int i = b * PB + j;
    int t = order[i >> 1], k = i & 1;
    int sl = slotarr[j];
    tok_slot[t * 2 + k] = sl;
    tok_gate[t * 2 + k] = (sl >= 0) ? wts[t * 2 + k] : 0.f;
    if (sl >= 0) slot_token[sl] = t;
  }
}

// ---------------------------------------------------------------------------
// 4. Gather tokens into per-expert bf16 A buffer (zero-fill empty slots).
//    One wave per slot row: float4 in, packed-uint2 (4 x bf16) out.
// ---------------------------------------------------------------------------
__global__ __launch_bounds__(32)
void k_gather(const int* __restrict__ slot_token, const float* __restrict__ xf,
              unsigned short* __restrict__ A) {
  int s = blockIdx.x, tid = threadIdx.x;       // tid = 0..31, 4 channels each
  int t = slot_token[s];
  uint2* row = (uint2*)(A + (size_t)s * Cc) + tid;
  if (t >= 0) {
    float4 v = *((const float4*)(xf + (size_t)t * Cc) + tid);
    uint2 pk;
    pk.x = (unsigned)f2bf(v.x) | ((unsigned)f2bf(v.y) << 16);
    pk.y = (unsigned)f2bf(v.z) | ((unsigned)f2bf(v.w) << 16);
    *row = pk;
  } else {
    *row = make_uint2(0u, 0u);
  }
}

// ---------------------------------------------------------------------------
// 5. Weight prep: fp32 -> bf16, transposed n-major for dword fragment loads.
//    W1 (E,C,MC) -> W1t[e][f][c];  W2 (E,MC,C) -> W2t[e][d][f]
// ---------------------------------------------------------------------------
__global__ void k_prepw(const float* __restrict__ W1, const float* __restrict__ W2,
                        unsigned short* __restrict__ W1t, unsigned short* __restrict__ W2t) {
  int idx = blockIdx.x * blockDim.x + threadIdx.x;
  const int tot = E * Cc * MC;
  if (idx < tot) {
    int e = idx / (Cc * MC), r = idx % (Cc * MC), c = r / MC, f = r % MC;
    W1t[((size_t)e * MC + f) * Cc + c] = f2bf(W1[idx]);
    int e2 = idx / (MC * Cc), r2 = idx % (MC * Cc), fi = r2 / Cc, d = r2 % Cc;
    W2t[((size_t)e2 * Cc + d) * MC + fi] = f2bf(W2[idx]);
  }
}

// ---------------------------------------------------------------------------
// 6. Fused expert MLP: Y = (gelu(A*W1 + b1)) * W2 + b2, H tile kept in LDS.
//    Block = 8 waves, 32 rows of one expert. 64 WMMAs per wave per block.
// ---------------------------------------------------------------------------
__global__ __launch_bounds__(256)
void k_expert(const unsigned short* __restrict__ A, const unsigned short* __restrict__ W1t,
              const unsigned short* __restrict__ W2t, const float* __restrict__ b1,
              const float* __restrict__ b2, float* __restrict__ Y) {
  __shared__ unsigned short Hlds[32 * HP];
  int e = blockIdx.y;
  int rowbase = blockIdx.x * 32;
  int lane = threadIdx.x & 31, wv = threadIdx.x >> 5;
  int nlane = lane & 15;
  int mhalf = (lane >> 4) << 3;
  const unsigned short* Ae  = A   + (size_t)e * CAP * Cc;
  const unsigned short* W1e = W1t + (size_t)e * MC * Cc;
  const unsigned short* W2e = W2t + (size_t)e * Cc * MC;
  int nb2 = wv * 16;

  // Warm the W2 rows this wave needs in stage 2 (global_prefetch_b8).
  __builtin_prefetch(W2e + (size_t)(nb2 + nlane) * MC, 0, 1);
  __builtin_prefetch(W2e + (size_t)(nb2 + nlane) * MC + 256, 0, 1);

  // ---- Stage 1: H columns [wv*64, wv*64+64), K = 128 ----
  int nb = wv * 64;
  for (int mt = 0; mt < 2; ++mt) {
    v16bf af[4];
#pragma unroll
    for (int ks = 0; ks < 4; ++ks)
      af[ks] = load_frag(Ae, rowbase + mt * 16, Cc, ks * 32, lane);
#pragma unroll
    for (int nt = 0; nt < 4; ++nt) {
      float bias = b1[(size_t)e * MC + nb + nt * 16 + nlane];
      v8f acc;
#pragma unroll
      for (int q = 0; q < 8; ++q) acc[q] = bias;
#pragma unroll
      for (int ks = 0; ks < 4; ++ks) {
        v16bf bf = load_frag(W1e, nb + nt * 16, Cc, ks * 32, lane);
        acc = __builtin_amdgcn_wmma_f32_16x16x32_bf16(
            false, af[ks], false, bf, (short)0, acc, false, false);
      }
#pragma unroll
      for (int q = 0; q < 8; ++q) {            // branchless GELU, bf16 to LDS
        float g = fast_gelu(acc[q]);
        int mr = mt * 16 + q + mhalf;
        Hlds[mr * HP + nb + nt * 16 + nlane] = f2bf(g);
      }
    }
  }
  __syncthreads();

  // ---- Stage 2: Y columns [wv*16, wv*16+16), K = 512 from LDS ----
  for (int mt = 0; mt < 2; ++mt) {
    float bias = b2[(size_t)e * Cc + nb2 + nlane];
    v8f acc;
#pragma unroll
    for (int q = 0; q < 8; ++q) acc[q] = bias;
    for (int ks = 0; ks < 16; ++ks) {
      v16bf afr = load_frag(Hlds, mt * 16, HP, ks * 32, lane);
      v16bf bfr = load_frag(W2e, nb2, MC, ks * 32, lane);
      acc = __builtin_amdgcn_wmma_f32_16x16x32_bf16(
          false, afr, false, bfr, (short)0, acc, false, false);
    }
#pragma unroll
    for (int q = 0; q < 8; ++q) {
      int mr = mt * 16 + q + mhalf;
      Y[((size_t)e * CAP + rowbase + mr) * Cc + nb2 + nlane] = acc[q];
    }
  }
}

// ---------------------------------------------------------------------------
// 7. Combine: out = input + x_skip + layer_scale * moe  (NCHW, coalesced in w)
// ---------------------------------------------------------------------------
__global__ __launch_bounds__(64)
void k_combine(const float* __restrict__ in, const float* __restrict__ xf,
               const float* __restrict__ Y, const int* __restrict__ tok_slot,
               const float* __restrict__ tok_gate, const float* __restrict__ ls,
               float* __restrict__ out) {
  int h = blockIdx.x, n = blockIdx.y, w = threadIdx.x;
  if (w >= Ww) return;
  int t = (n * Hh + h) * Ww + w;
  int s0 = tok_slot[t * 2], s1 = tok_slot[t * 2 + 1];
  float g0 = tok_gate[t * 2], g1 = tok_gate[t * 2 + 1];
  const float4* y0 = (s0 >= 0) ? (const float4*)(Y + (size_t)s0 * Cc) : nullptr;
  const float4* y1 = (s1 >= 0) ? (const float4*)(Y + (size_t)s1 * Cc) : nullptr;
  const float4* xs = (const float4*)(xf + (size_t)t * Cc);
  for (int c4 = 0; c4 < Cc / 4; ++c4) {
    float4 m = make_float4(0.f, 0.f, 0.f, 0.f);
    if (y0) { float4 a = y0[c4]; m.x += a.x * g0; m.y += a.y * g0; m.z += a.z * g0; m.w += a.w * g0; }
    if (y1) { float4 a = y1[c4]; m.x += a.x * g1; m.y += a.y * g1; m.z += a.z * g1; m.w += a.w * g1; }
    float4 xv = xs[c4];
    float mm[4] = {m.x, m.y, m.z, m.w};
    float xx[4] = {xv.x, xv.y, xv.z, xv.w};
#pragma unroll
    for (int j = 0; j < 4; ++j) {
      int c = c4 * 4 + j;
      size_t o = ((size_t)(n * Cc + c) * Hh + h) * Ww + w;
      out[o] = in[o] + xx[j] + ls[c] * mm[j];
    }
  }
}

// ---------------------------------------------------------------------------
extern "C" void kernel_launch(void* const* d_in, const int* in_sizes, int n_in,
                              void* d_out, int out_size, void* d_ws, size_t ws_size,
                              hipStream_t stream) {
  const float* in  = (const float*)d_in[0];
  const float* wk  = (const float*)d_in[1];
  const float* wb  = (const float*)d_in[2];
  const float* gam = (const float*)d_in[3];
  const float* bet = (const float*)d_in[4];
  const float* Wg  = (const float*)d_in[5];
  const float* bg  = (const float*)d_in[6];
  const float* W1  = (const float*)d_in[7];
  const float* b1  = (const float*)d_in[8];
  const float* W2  = (const float*)d_in[9];
  const float* b2  = (const float*)d_in[10];
  const float* ls  = (const float*)d_in[11];
  float* out = (float*)d_out;

  char* p = (char*)d_ws;
  auto alloc = [&](size_t bytes) -> void* {
    void* r = (void*)p;
    p += (bytes + 255) & ~(size_t)255;
    return r;
  };
  float*          xf         = (float*)alloc(sizeof(float) * (size_t)T * Cc);
  float*          prio       = (float*)alloc(sizeof(float) * T);
  int*            topi       = (int*)alloc(sizeof(int) * T * 2);
  float*          wts        = (float*)alloc(sizeof(float) * T * 2);
  unsigned*       hist       = (unsigned*)alloc(sizeof(unsigned) * 65536);
  int*            order      = (int*)alloc(sizeof(int) * T);
  unsigned*       bcnt       = (unsigned*)alloc(sizeof(unsigned) * NBLK * E);
  unsigned*       boff       = (unsigned*)alloc(sizeof(unsigned) * NBLK * E);
  int*            tok_slot   = (int*)alloc(sizeof(int) * T * 2);
  float*          tok_gate   = (float*)alloc(sizeof(float) * T * 2);
  int*            slot_token = (int*)alloc(sizeof(int) * ECAP);
  unsigned short* Abuf       = (unsigned short*)alloc(sizeof(unsigned short) * (size_t)ECAP * Cc);
  float*          Ybuf       = (float*)alloc(sizeof(float) * (size_t)ECAP * Cc);
  unsigned short* W1t        = (unsigned short*)alloc(sizeof(unsigned short) * (size_t)E * MC * Cc);
  unsigned short* W2t        = (unsigned short*)alloc(sizeof(unsigned short) * (size_t)E * Cc * MC);

  hipMemsetAsync(hist, 0, sizeof(unsigned) * 65536, stream);
  hipMemsetAsync(slot_token, 0xFF, sizeof(int) * ECAP, stream);

  k_prepw<<<(E * Cc * MC + 255) / 256, 256, 0, stream>>>(W1, W2, W1t, W2t);
  k_dwln_gate<<<dim3(Hh, Nn), 64, 0, stream>>>(in, wk, wb, gam, bet, Wg, bg,
                                               xf, prio, topi, wts);
  k_hist<<<(T + 255) / 256, 256, 0, stream>>>(prio, hist);
  k_scan<<<1, 256, 0, stream>>>(hist);
  k_scatter<<<(T + 255) / 256, 256, 0, stream>>>(prio, hist, order);
  k_blockcnt<<<NBLK, 256, 0, stream>>>(order, topi, bcnt);
  k_blockscan<<<1, 8, 0, stream>>>(bcnt, boff);
  k_positions<<<NBLK, 256, 0, stream>>>(order, topi, wts, boff,
                                        tok_slot, tok_gate, slot_token);
  k_gather<<<ECAP, 32, 0, stream>>>(slot_token, xf, Abuf);
  k_expert<<<dim3(CAP / 32, E), 256, 0, stream>>>(Abuf, W1t, W2t, b1, b2, Ybuf);
  k_combine<<<dim3(Hh, Nn), 64, 0, stream>>>(in, xf, Ybuf, tok_slot, tok_gate, ls, out);
}
